// MoE_52536039965044
// MI455X (gfx1250) — matrix operand order repaired and checked by reading, not compiled, not run on hardware
//
#include <hip/hip_runtime.h>

// ---------------------------------------------------------------------------
// MoE (DeepSeek-style) for MI455X / gfx1250, wave32 + WMMA bf16 + TDM.
// Routed experts (dense, combine-scaled) + shared expert folded into two big
// GEMMs using split-bf16 (hi+lo) V_WMMA_F32_16X16X32_BF16, fp32 accumulate
// (3 WMMAs per fp32 product: hi*hi + hi*lo + lo*hi ~ fp32 precision).
// B operands double-buffered through LDS via the Tensor Data Mover
// (tensor_load_to_lds, TENSORcnt) -- fallback: cooperative b128 copies.
// ---------------------------------------------------------------------------

typedef __attribute__((ext_vector_type(16))) __bf16 bf16x16;
typedef __attribute__((ext_vector_type(8)))  __bf16 bf16x8;
typedef __attribute__((ext_vector_type(8)))  float  v8f;

#define TOK   8192      // B*S tokens
#define DH    1024      // d_hidden
#define NEXP  8
#define DEXP  256
#define SDE   512
#define NP1   5120      // packed B1 columns: 2*(NEXP*DEXP + SDE)
#define NH    2560      // h columns: NEXP*DEXP + SDE
#define LDH   2560

// ------------------------- TDM availability --------------------------------
#if defined(__has_builtin)
#  if __has_builtin(__builtin_amdgcn_tensor_load_to_lds)
#    define CDNA5_USE_TDM 1
#    if __has_include(<hip/amd_detail/amd_gfx1250_TDM.h>)
#      define CDNA5_TDM_6ARG 1
#    endif
#  endif
#endif

#ifdef CDNA5_USE_TDM
typedef __attribute__((ext_vector_type(4))) unsigned int v4u;
typedef __attribute__((ext_vector_type(8))) int          v8i;
typedef __attribute__((ext_vector_type(4))) int          v4i;

// 2D TDM load: tile = 2 rows x 4096B (two k16-groups of a 128-col B panel),
// row stride = stride_elems (bf16 elements), dest = contiguous 8KB in LDS.
// D# packing per CDNA5 ISA 8.3/8.4 (group0: count/lds/global/type,
// group1: data_size=2B, tensor dims, tile dims, dim0 stride).
__device__ __forceinline__ void tdm_load_b_panel(const __bf16* gsrc,
                                                 unsigned lds_off,
                                                 unsigned stride_elems)
{
    unsigned long long ga = (unsigned long long)(uintptr_t)gsrc;
    v4u g0;
    g0[0] = 1u;                                      // count=1, user desc
    g0[1] = lds_off;                                 // LDS byte address
    g0[2] = (unsigned)(ga & 0xFFFFFFFFu);            // global_addr[31:0]
    g0[3] = (unsigned)((ga >> 32) & 0x01FFFFFFu)     // global_addr[56:32]
          | (2u << 30);                              // type = 2 ("image")
    v8i g1;
    g1[0] = (int)(1u << 16);                         // data_size = 1 (2 bytes)
    g1[1] = (int)(2048u << 16);                      // tensor_dim0 = 2048 elems
    g1[2] = (int)(64u << 16);                        // tensor_dim1 = 64 rows
    g1[3] = (int)(2048u << 16);                      // tile_dim0 = 2048 elems
    g1[4] = 2;                                       // tile_dim1 = 2 rows
    g1[5] = (int)stride_elems;                       // tensor_dim0_stride lo32
    g1[6] = 0;                                       // stride hi16 | dim1_stride
    g1[7] = 0;
    v4i z4 = {0, 0, 0, 0};
#ifdef CDNA5_TDM_6ARG
    v8i z8 = {0, 0, 0, 0, 0, 0, 0, 0};
    __builtin_amdgcn_tensor_load_to_lds(g0, g1, z4, z4, z8, 0);
#else
    __builtin_amdgcn_tensor_load_to_lds(g0, g1, z4, z4, 0);
#endif
}

__device__ __forceinline__ void tdm_wait0()
{
#if __has_builtin(__builtin_amdgcn_s_wait_tensorcnt)
    __builtin_amdgcn_s_wait_tensorcnt((short)0);
#else
    asm volatile("s_wait_tensorcnt 0x0" ::: "memory");
#endif
}
#endif  // CDNA5_USE_TDM

// ---------------------------- router ---------------------------------------
__global__ __launch_bounds__(128) void router_kernel(
    const float* __restrict__ x, const float* __restrict__ Wg,
    float* __restrict__ combine)
{
    const int lane = threadIdx.x & 31;
    const int wid  = threadIdx.x >> 5;
    const int t    = blockIdx.x * 4 + wid;

    float acc[NEXP];
#pragma unroll
    for (int e = 0; e < NEXP; ++e) acc[e] = 0.0f;

    const float* xr = x + (size_t)t * DH;
    for (int d = lane; d < DH; d += 32) {
        float xv = xr[d];
        const float* wr = Wg + (size_t)d * NEXP;
#pragma unroll
        for (int e = 0; e < NEXP; ++e) acc[e] += xv * wr[e];
    }
#pragma unroll
    for (int off = 16; off > 0; off >>= 1) {
#pragma unroll
        for (int e = 0; e < NEXP; ++e)
            acc[e] += __shfl_xor(acc[e], off, 32);
    }
    if (lane == 0) {
        float m = acc[0];
#pragma unroll
        for (int e = 1; e < NEXP; ++e) m = fmaxf(m, acc[e]);
        float p[NEXP], sum = 0.0f;
#pragma unroll
        for (int e = 0; e < NEXP; ++e) { p[e] = __expf(acc[e] - m); sum += p[e]; }
        float inv = 1.0f / sum;
#pragma unroll
        for (int e = 0; e < NEXP; ++e) p[e] *= inv;
        int i1 = 0;
#pragma unroll
        for (int e = 1; e < NEXP; ++e) if (p[e] > p[i1]) i1 = e;
        int i2 = (i1 == 0) ? 1 : 0;
#pragma unroll
        for (int e = 0; e < NEXP; ++e) if (e != i1 && p[e] > p[i2]) i2 = e;
        float* c = combine + (size_t)t * NEXP;
#pragma unroll
        for (int e = 0; e < NEXP; ++e)
            c[e] = (e == i1 || e == i2) ? p[e] : 0.0f;
    }
}

// ------------------------- fp32 -> bf16 hi/lo -------------------------------
__global__ __launch_bounds__(256) void split_x_kernel(
    const float* __restrict__ src, __bf16* __restrict__ hi,
    __bf16* __restrict__ lo)
{
    size_t i = (size_t)blockIdx.x * 256 + threadIdx.x;
    float v = src[i];
    __bf16 h = (__bf16)v;
    hi[i] = h;
    lo[i] = (__bf16)(v - (float)h);
}

// --------------------- pack B1 (gate/up, routed+shared) ---------------------
__global__ __launch_bounds__(256) void pack_b1_kernel(
    const float* __restrict__ w_gate, const float* __restrict__ w_up,
    const float* __restrict__ sg, const float* __restrict__ su,
    __bf16* __restrict__ b1h, __bf16* __restrict__ b1l)
{
    size_t idx = (size_t)blockIdx.x * 256 + threadIdx.x;   // < DH*NP1
    int k = (int)(idx / NP1);
    int p = (int)(idx - (size_t)k * NP1);
    int j = p >> 5, i5 = p & 31;
    int up = (i5 >> 4) & 1, i = i5 & 15;
    float v;
    if (j < 128) {
        int e = j >> 4;
        int c = ((j & 15) << 4) + i;
        const float* s = up ? w_up : w_gate;
        v = s[((size_t)e * DH + k) * DEXP + c];
    } else {
        int c = ((j - 128) << 4) + i;
        v = (up ? su : sg)[(size_t)k * SDE + c];
    }
    __bf16 h = (__bf16)v;
    size_t o = ((size_t)(k >> 4) * NP1 + p) * 16 + (k & 15);
    b1h[o] = h;
    b1l[o] = (__bf16)(v - (float)h);
}

// --------------------- pack B2 ([w_down_flat ; sd]) -------------------------
__global__ __launch_bounds__(256) void pack_b2_kernel(
    const float* __restrict__ w_down, const float* __restrict__ sd,
    __bf16* __restrict__ b2h, __bf16* __restrict__ b2l)
{
    size_t idx = (size_t)blockIdx.x * 256 + threadIdx.x;   // < NH*DH
    int k = (int)(idx >> 10);
    int n = (int)(idx & 1023);
    float v = (k < NEXP * DEXP) ? w_down[(size_t)k * DH + n]
                                : sd[(size_t)(k - NEXP * DEXP) * DH + n];
    __bf16 h = (__bf16)v;
    size_t o = ((size_t)(k >> 4) * DH + n) * 16 + (k & 15);
    b2h[o] = h;
    b2l[o] = (__bf16)(v - (float)h);
}

// ------------------------- WMMA fragment helpers ----------------------------
__device__ __forceinline__ bf16x16 load_a_frag(const __bf16* A, int lda,
                                               int row0, int kb, int lane)
{
    int lr = lane & 15, lh = lane >> 4;
    const __bf16* p = A + (size_t)(row0 + lr) * lda + kb + lh * 16;
    bf16x8 c0 = *(const bf16x8*)p;
    bf16x8 c1 = *(const bf16x8*)(p + 8);
    bf16x16 a;
#pragma unroll
    for (int q = 0; q < 8; ++q) { a[q] = c0[q]; a[q + 8] = c1[q]; }
    return a;
}

__device__ __forceinline__ v8f wmma3(bf16x16 ah, bf16x16 al,
                                     bf16x16 bh, bf16x16 bl, v8f c)
{
    c = __builtin_amdgcn_wmma_f32_16x16x32_bf16(false, ah, false, bh, (short)0, c, false, false);
    c = __builtin_amdgcn_wmma_f32_16x16x32_bf16(false, ah, false, bl, (short)0, c, false, false);
    c = __builtin_amdgcn_wmma_f32_16x16x32_bf16(false, al, false, bh, (short)0, c, false, false);
    return c;
}

// ----------- fallback cooperative B stage: global -> LDS --------------------
template<int LDN>
__device__ __forceinline__ void stage_b(const __bf16* __restrict__ bh,
                                        const __bf16* __restrict__ bl,
                                        int kb, int nblk,
                                        __bf16* dst, int tid)
{
#pragma unroll
    for (int q = 0; q < 2; ++q) {
        int c    = tid + q * 256;        // 0..511 : 32B chunk id
        int part = c >> 7;               // 0..3
        int col  = c & 127;
        int kg   = part & 1;
        const __bf16* src = ((part < 2) ? bh : bl)
            + ((size_t)((kb >> 4) + kg) * LDN + (nblk + col)) * 16;
        *(bf16x16*)(dst + (((part << 7) + col) << 4)) = *(const bf16x16*)src;
    }
}

__device__ __forceinline__ bf16x16 lds_b_frag(const __bf16* base, int col, int lh)
{
    return *(const bf16x16*)(base + (((lh << 7) + col) << 4));
}

// ------------------------------- GEMM1 --------------------------------------
// h[T,2560] = swiglu(x @ B1) * combine, split to bf16 hi/lo.
// Block: 8 waves (4 M x 2 N), tile 128 rows x 128 packed cols.
__global__ __launch_bounds__(256) void gemm1_kernel(
    const __bf16* __restrict__ xh, const __bf16* __restrict__ xl,
    const __bf16* __restrict__ b1h, const __bf16* __restrict__ b1l,
    const float* __restrict__ combine,
    __bf16* __restrict__ hh, __bf16* __restrict__ hl)
{
    __shared__ __align__(64) __bf16 Bs[2][4 * 128 * 16];   // 32KB double buffer

    const int tid  = threadIdx.x;
    const int lane = tid & 31, wid = tid >> 5;
    const int lc = lane & 15, lh = lane >> 4;
    const int m0   = blockIdx.y * 128 + (wid & 3) * 32;
    const int nblk = blockIdx.x * 128;                     // packed col base
    const int nw   = (wid >> 2) * 64;                      // wave col offset

    v8f acc[2][4] = {};

#ifdef CDNA5_USE_TDM
    const unsigned ldsb0 = (unsigned)(uintptr_t)(void*)&Bs[0][0];
    const unsigned ldsb1 = (unsigned)(uintptr_t)(void*)&Bs[1][0];
    if (wid == 0) {
        const __bf16* srch = b1h + (size_t)nblk * 16;
        const __bf16* srcl = b1l + (size_t)nblk * 16;
        tdm_load_b_panel(srch, ldsb0,        NP1 * 16);
        tdm_load_b_panel(srcl, ldsb0 + 8192, NP1 * 16);
        tdm_wait0();
    }
#else
    stage_b<NP1>(b1h, b1l, 0, nblk, &Bs[0][0], tid);
#endif
    __syncthreads();

    int buf = 0;
    for (int ks = 0; ks < DH / 32; ++ks) {
        const int kb = ks * 32;
#ifdef CDNA5_USE_TDM
        if (ks + 1 < DH / 32 && wid == 0) {
            size_t roff = ((size_t)((kb + 32) >> 4) * NP1 + nblk) * 16;
            unsigned dst = (buf == 0) ? ldsb1 : ldsb0;
            tdm_load_b_panel(b1h + roff, dst,        NP1 * 16);
            tdm_load_b_panel(b1l + roff, dst + 8192, NP1 * 16);
        }
#else
        if (ks + 1 < DH / 32)
            stage_b<NP1>(b1h, b1l, kb + 32, nblk, &Bs[buf ^ 1][0], tid);
#endif
        if (kb + 128 < DH) {   // prefetch A four steps ahead
            __builtin_prefetch(xh + (size_t)(m0 + lane) * DH + kb + 128, 0, 0);
            __builtin_prefetch(xl + (size_t)(m0 + lane) * DH + kb + 128, 0, 0);
        }
        bf16x16 aH0 = load_a_frag(xh, DH, m0,      kb, lane);
        bf16x16 aH1 = load_a_frag(xh, DH, m0 + 16, kb, lane);
        bf16x16 aL0 = load_a_frag(xl, DH, m0,      kb, lane);
        bf16x16 aL1 = load_a_frag(xl, DH, m0 + 16, kb, lane);
#pragma unroll
        for (int nt = 0; nt < 4; ++nt) {
            const __bf16* bp = &Bs[buf][0];
            bf16x16 bH = lds_b_frag(bp,                 nw + nt * 16 + lc, lh);
            bf16x16 bL = lds_b_frag(bp + 2 * 128 * 16,  nw + nt * 16 + lc, lh);
            acc[0][nt] = wmma3(aH0, aL0, bH, bL, acc[0][nt]);
            acc[1][nt] = wmma3(aH1, aL1, bH, bL, acc[1][nt]);
        }
#ifdef CDNA5_USE_TDM
        if (wid == 0) tdm_wait0();
#endif
        __syncthreads();
        buf ^= 1;
    }

    const int p0 = nblk + nw;          // multiple of 64
#pragma unroll
    for (int pair = 0; pair < 2; ++pair) {
        const int j = (p0 >> 5) + pair;        // h 16-col group, 0..159
        const int e = j >> 4;
        const bool routed = (j < 128);
        const int col = j * 16 + lc;
#pragma unroll
        for (int mt = 0; mt < 2; ++mt) {
#pragma unroll
            for (int r = 0; r < 8; ++r) {
                int row = m0 + mt * 16 + lh * 8 + r;
                float g = acc[mt][2 * pair][r];
                float u = acc[mt][2 * pair + 1][r];
                float s = routed ? combine[(size_t)row * NEXP + e] : 1.0f;
                float hv = g * (1.0f / (1.0f + __expf(-g))) * u * s;
                __bf16 hb = (__bf16)hv;
                size_t o = (size_t)row * LDH + col;
                hh[o] = hb;
                hl[o] = (__bf16)(hv - (float)hb);
            }
        }
    }
}

// ------------------------------- GEMM2 --------------------------------------
// out[T,1024] = h @ [w_down_flat ; sd]  (routed-scaled + shared in one GEMM)
__global__ __launch_bounds__(256) void gemm2_kernel(
    const __bf16* __restrict__ hh, const __bf16* __restrict__ hl,
    const __bf16* __restrict__ b2h, const __bf16* __restrict__ b2l,
    float* __restrict__ out)
{
    __shared__ __align__(64) __bf16 Bs[2][4 * 128 * 16];   // 32KB double buffer

    const int tid  = threadIdx.x;
    const int lane = tid & 31, wid = tid >> 5;
    const int lc = lane & 15, lh = lane >> 4;
    const int m0   = blockIdx.y * 128 + (wid & 3) * 32;
    const int nblk = blockIdx.x * 128;
    const int nw   = (wid >> 2) * 64;

    v8f acc[2][4] = {};

#ifdef CDNA5_USE_TDM
    const unsigned ldsb0 = (unsigned)(uintptr_t)(void*)&Bs[0][0];
    const unsigned ldsb1 = (unsigned)(uintptr_t)(void*)&Bs[1][0];
    if (wid == 0) {
        tdm_load_b_panel(b2h + (size_t)nblk * 16, ldsb0,        DH * 16);
        tdm_load_b_panel(b2l + (size_t)nblk * 16, ldsb0 + 8192, DH * 16);
        tdm_wait0();
    }
#else
    stage_b<DH>(b2h, b2l, 0, nblk, &Bs[0][0], tid);
#endif
    __syncthreads();

    int buf = 0;
    for (int ks = 0; ks < NH / 32; ++ks) {
        const int kb = ks * 32;
#ifdef CDNA5_USE_TDM
        if (ks + 1 < NH / 32 && wid == 0) {
            size_t roff = ((size_t)((kb + 32) >> 4) * DH + nblk) * 16;
            unsigned dst = (buf == 0) ? ldsb1 : ldsb0;
            tdm_load_b_panel(b2h + roff, dst,        DH * 16);
            tdm_load_b_panel(b2l + roff, dst + 8192, DH * 16);
        }
#else
        if (ks + 1 < NH / 32)
            stage_b<DH>(b2h, b2l, kb + 32, nblk, &Bs[buf ^ 1][0], tid);
#endif
        if (kb + 128 < NH) {
            __builtin_prefetch(hh + (size_t)(m0 + lane) * LDH + kb + 128, 0, 0);
            __builtin_prefetch(hl + (size_t)(m0 + lane) * LDH + kb + 128, 0, 0);
        }
        bf16x16 aH0 = load_a_frag(hh, LDH, m0,      kb, lane);
        bf16x16 aH1 = load_a_frag(hh, LDH, m0 + 16, kb, lane);
        bf16x16 aL0 = load_a_frag(hl, LDH, m0,      kb, lane);
        bf16x16 aL1 = load_a_frag(hl, LDH, m0 + 16, kb, lane);
#pragma unroll
        for (int nt = 0; nt < 4; ++nt) {
            const __bf16* bp = &Bs[buf][0];
            bf16x16 bH = lds_b_frag(bp,                nw + nt * 16 + lc, lh);
            bf16x16 bL = lds_b_frag(bp + 2 * 128 * 16, nw + nt * 16 + lc, lh);
            acc[0][nt] = wmma3(aH0, aL0, bH, bL, acc[0][nt]);
            acc[1][nt] = wmma3(aH1, aL1, bH, bL, acc[1][nt]);
        }
#ifdef CDNA5_USE_TDM
        if (wid == 0) tdm_wait0();
#endif
        __syncthreads();
        buf ^= 1;
    }

#pragma unroll
    for (int nt = 0; nt < 4; ++nt) {
#pragma unroll
        for (int mt = 0; mt < 2; ++mt) {
#pragma unroll
            for (int r = 0; r < 8; ++r) {
                int row = m0 + mt * 16 + lh * 8 + r;
                int col = nblk + nw + nt * 16 + lc;
                out[(size_t)row * DH + col] = acc[mt][nt][r];
            }
        }
    }
}

// ------------------------------- launcher -----------------------------------
extern "C" void kernel_launch(void* const* d_in, const int* in_sizes, int n_in,
                              void* d_out, int out_size, void* d_ws, size_t ws_size,
                              hipStream_t stream)
{
    (void)in_sizes; (void)n_in; (void)out_size; (void)ws_size;

    const float* x      = (const float*)d_in[0];
    const float* Wg     = (const float*)d_in[1];
    const float* w_gate = (const float*)d_in[2];
    const float* w_up   = (const float*)d_in[3];
    const float* w_down = (const float*)d_in[4];
    const float* sg     = (const float*)d_in[5];
    const float* su     = (const float*)d_in[6];
    const float* sd     = (const float*)d_in[7];
    // d_in[8] = top_k (hardcoded K=2)
    float* out = (float*)d_out;

    char* ws = (char*)d_ws;
    __bf16* x_hi = (__bf16*)ws;  ws += (size_t)TOK * DH * 2;
    __bf16* x_lo = (__bf16*)ws;  ws += (size_t)TOK * DH * 2;
    __bf16* b1h  = (__bf16*)ws;  ws += (size_t)DH * NP1 * 2;
    __bf16* b1l  = (__bf16*)ws;  ws += (size_t)DH * NP1 * 2;
    __bf16* b2h  = (__bf16*)ws;  ws += (size_t)NH * DH * 2;
    __bf16* b2l  = (__bf16*)ws;  ws += (size_t)NH * DH * 2;
    __bf16* h_hi = (__bf16*)ws;  ws += (size_t)TOK * NH * 2;
    __bf16* h_lo = (__bf16*)ws;  ws += (size_t)TOK * NH * 2;
    float*  comb = (float*)ws;   ws += (size_t)TOK * NEXP * 4;

    router_kernel<<<TOK / 4, 128, 0, stream>>>(x, Wg, comb);
    split_x_kernel<<<(TOK * DH) / 256, 256, 0, stream>>>(x, x_hi, x_lo);
    pack_b1_kernel<<<(DH * NP1) / 256, 256, 0, stream>>>(w_gate, w_up, sg, su, b1h, b1l);
    pack_b2_kernel<<<(NH * DH) / 256, 256, 0, stream>>>(w_down, sd, b2h, b2l);
    gemm1_kernel<<<dim3(NP1 / 128, TOK / 128), 256, 0, stream>>>(
        x_hi, x_lo, b1h, b1l, comb, h_hi, h_lo);
    gemm2_kernel<<<dim3(DH / 128, TOK / 128), 256, 0, stream>>>(
        h_hi, h_lo, b2h, b2l, out);
}
